// GNNPrediction_34050500722852
// MI455X (gfx1250) — compile-verified, matrix operand order checked by reading, not run on hardware
//
#include <hip/hip_runtime.h>
#include <hip/hip_bf16.h>

typedef __attribute__((ext_vector_type(16))) __bf16 v16bf;
typedef __attribute__((ext_vector_type(8)))  float  v8f;

#define NMAX       128
#define D_NODE     96
#define LATENT     512
#define OUT_DIM    601
#define NPAIR      8128     // 128*127/2
#define POLICY_DIM 8129
#define HIDDEN     128
#define WSB_DWORDS (12 * 8 * 32 * 8)   // 24576 dwords = 98304 bytes

// ---------------------------------------------------------------------------
// bf16 round-to-nearest-even from f32
// ---------------------------------------------------------------------------
__device__ __forceinline__ unsigned short f2bf_rne(float f) {
    union { float f; unsigned u; } v; v.f = f;
    unsigned u = v.u;
    u += 0x7FFFu + ((u >> 16) & 1u);
    return (unsigned short)(u >> 16);
}

// ---------------------------------------------------------------------------
// Pack sW1 (f32, [384][128] row-major fan_in x fan_out) into bf16 WMMA
// B-fragments. Fragment (kk, nt): 32 lanes x 8 dwords (2 bf16 each).
//   lane < 16 : N = nt*16+lane,  K-half offset 0
//   lane >= 16: N = nt*16+lane-16, K-half offset 8
//   dword r<4 : K = kk*32 + off + 2r {+0,+1}
//   dword r>=4: K = kk*32 + 16 + off + 2(r-4) {+0,+1}
// Layout: wsB[((kk*8+nt)*32 + lane)*8 + r]
// ---------------------------------------------------------------------------
__global__ void pack_w1_kernel(const float* __restrict__ sW1,
                               unsigned* __restrict__ wsB) {
    int idx = blockIdx.x * 256 + threadIdx.x;
    if (idx >= WSB_DWORDS) return;
    int r    = idx & 7;
    int lane = (idx >> 3) & 31;
    int nt   = (idx >> 8) & 7;
    int kk   = idx >> 11;
    int n    = nt * 16 + (lane & 15);
    int off  = (lane >= 16) ? 8 : 0;
    int krel = (r < 4) ? (off + r * 2) : (16 + off + (r - 4) * 2);
    int k0   = kk * 32 + krel;
    unsigned lo = f2bf_rne(sW1[(size_t)k0 * HIDDEN + n]);
    unsigned hi = f2bf_rne(sW1[(size_t)(k0 + 1) * HIDDEN + n]);
    wsB[idx] = lo | (hi << 16);
}

// ---------------------------------------------------------------------------
// triu_indices(128, k=1) tables
// ---------------------------------------------------------------------------
__global__ void pair_table_kernel(int* __restrict__ ii, int* __restrict__ jj) {
    int p = blockIdx.x * 256 + threadIdx.x;
    if (p >= NPAIR) return;
    int i = 0, rem = p, rowlen = NMAX - 1;
    while (rem >= rowlen) { rem -= rowlen; rowlen--; i++; }
    ii[p] = i;
    jj[p] = i + 1 + rem;
}

// ---------------------------------------------------------------------------
// Edge MLP: one wave per 16-pair tile, 4 waves/block.
//   feat(16x384) @ sW1(384x128) -> relu -> @ sW2(128x1)
// Packed bf16 weights are staged global->LDS once per block with
// gfx1250 async copies (ASYNCcnt), then each WMMA's B fragment is two
// conflict-free ds_load_b128.
// ---------------------------------------------------------------------------
__global__ __launch_bounds__(128, 1)
void edge_mlp_kernel(const float*    __restrict__ H,
                     const unsigned* __restrict__ wsB,
                     const int*      __restrict__ ii,
                     const int*      __restrict__ jj,
                     const float*    __restrict__ sb1,
                     const float*    __restrict__ sW2,
                     const float*    __restrict__ sb2,
                     float*          __restrict__ policy) {
    __shared__ unsigned smemB[WSB_DWORDS];       // 96 KB packed bf16 sW1

    const int lane = threadIdx.x & 31;
    const int wave = threadIdx.x >> 5;
    const int tile = blockIdx.x * 4 + wave;      // 127*4 = 508 tiles exactly
    const int b    = blockIdx.y;
    const int p0   = tile * 16;
    const int ml   = lane & 15;                  // A-layout: M = lane & 15
    const int p    = p0 + ml;

    // --- async-stage weights into LDS (98304 B = 48 iters * 128 thr * 16 B) ---
    {
        const unsigned ldsBase = (unsigned)(size_t)(&smemB[0]); // LDS byte offset
        const unsigned long long gbase = (unsigned long long)wsB;
#pragma unroll 4
        for (int it = 0; it < 48; ++it) {
            unsigned off = (unsigned)(it * 2048 + threadIdx.x * 16);
            asm volatile("global_load_async_to_lds_b128 %0, %1, %2"
                         :: "v"(ldsBase + off), "v"(off), "s"(gbase)
                         : "memory");
        }
    }

    // overlap: per-lane pair setup while async copies are in flight
    const int ni = ii[p];
    const int nj = jj[p];
    const float* Hi = H + ((size_t)b * NMAX + ni) * D_NODE;
    const float* Hj = H + ((size_t)b * NMAX + nj) * D_NODE;

    v8f acc[8];
#pragma unroll
    for (int nt = 0; nt < 8; ++nt) acc[nt] = (v8f){0.f,0.f,0.f,0.f,0.f,0.f,0.f,0.f};

    const int off = (lane >= 16) ? 8 : 0;        // K-half per lane group

    asm volatile("s_wait_asynccnt 0x0" ::: "memory");
    __syncthreads();

#pragma unroll
    for (int dstep = 0; dstep < 3; ++dstep) {
        const int d0 = dstep * 32 + off;
        // this lane's 16 Hi / 16 Hj features for this d-step (b128 loads)
        float4 hiA = *(const float4*)(Hi + d0);
        float4 hiB = *(const float4*)(Hi + d0 + 4);
        float4 hiC = *(const float4*)(Hi + d0 + 16);
        float4 hiD = *(const float4*)(Hi + d0 + 20);
        float4 hjA = *(const float4*)(Hj + d0);
        float4 hjB = *(const float4*)(Hj + d0 + 4);
        float4 hjC = *(const float4*)(Hj + d0 + 16);
        float4 hjD = *(const float4*)(Hj + d0 + 20);

        float hv[16] = {hiA.x,hiA.y,hiA.z,hiA.w, hiB.x,hiB.y,hiB.z,hiB.w,
                        hiC.x,hiC.y,hiC.z,hiC.w, hiD.x,hiD.y,hiD.z,hiD.w};
        float jv[16] = {hjA.x,hjA.y,hjA.z,hjA.w, hjB.x,hjB.y,hjB.z,hjB.w,
                        hjC.x,hjC.y,hjC.z,hjC.w, hjD.x,hjD.y,hjD.z,hjD.w};

#pragma unroll
        for (int seg = 0; seg < 4; ++seg) {
            v16bf a;
#pragma unroll
            for (int e = 0; e < 16; ++e) {
                float x = hv[e], y = jv[e];
                float v = (seg == 0) ? x
                        : (seg == 1) ? y
                        : (seg == 2) ? fabsf(x - y)
                                     : x * y;
                a[e] = (__bf16)v;
            }
            const int kk = seg * 3 + dstep;
            const unsigned* bp = &smemB[((kk * 8) * 32 + lane) * 8];
#pragma unroll
            for (int nt = 0; nt < 8; ++nt) {
                union { v16bf v; uint4 q[2]; } bu;
                bu.q[0] = *(const uint4*)(bp + nt * 256);
                bu.q[1] = *(const uint4*)(bp + nt * 256 + 4);
                acc[nt] = __builtin_amdgcn_wmma_f32_16x16x32_bf16(
                    false, a, false, bu.v, (short)0, acc[nt], false, false);
            }
        }
    }

    // Epilogue: bias + ReLU, then hidden(16x128) @ sW2(128x1).
    // Lane holds column N = nt*16+ml; VGPR r holds row M=r (lanes<16) / r+8.
    float part[8];
#pragma unroll
    for (int r = 0; r < 8; ++r) part[r] = 0.f;
#pragma unroll
    for (int nt = 0; nt < 8; ++nt) {
        const int n  = nt * 16 + ml;
        const float bn = sb1[n];
        const float wn = sW2[n];
#pragma unroll
        for (int r = 0; r < 8; ++r) {
            float h = fmaxf(acc[nt][r] + bn, 0.f);
            part[r] = fmaf(h, wn, part[r]);
        }
    }
    // reduce over the 16 lanes sharing the same M-half (masks < 16 stay in half)
#pragma unroll
    for (int mask = 1; mask < 16; mask <<= 1) {
#pragma unroll
        for (int r = 0; r < 8; ++r)
            part[r] += __shfl_xor(part[r], mask, 32);
    }
    if (ml == 0) {
        const int mbase = (lane >= 16) ? 8 : 0;
        const float bias2 = sb2[0];
        float* dst = policy + (size_t)b * POLICY_DIM + p0 + mbase;
#pragma unroll
        for (int r = 0; r < 8; ++r) dst[r] = part[r] + bias2;
    }
}

// ---------------------------------------------------------------------------
// Value / reward heads (tiny; plain VALU)
// ---------------------------------------------------------------------------
__global__ void head_hidden_kernel(const float* __restrict__ z,
                                   const float* __restrict__ vW1, const float* __restrict__ vb1,
                                   const float* __restrict__ rW1, const float* __restrict__ rb1,
                                   float* __restrict__ wsH) {
    __shared__ float zs[LATENT];
    const int b     = blockIdx.x;
    const int which = blockIdx.y;                 // 0 = value, 1 = reward
    const float* W1 = which ? rW1 : vW1;
    const float* b1 = which ? rb1 : vb1;
    for (int k = threadIdx.x; k < LATENT; k += blockDim.x)
        zs[k] = z[(size_t)b * LATENT + k];
    __syncthreads();
    const int n = threadIdx.x;                    // blockDim = 256
    float s = b1[n];
    for (int k = 0; k < LATENT; ++k)
        s = fmaf(zs[k], W1[(size_t)k * 256 + n], s);
    wsH[((size_t)which * 64 + b) * 256 + n] = fmaxf(s, 0.f);
}

__global__ void head_out_kernel(const float* __restrict__ wsH,
                                const float* __restrict__ vW2, const float* __restrict__ vb2,
                                const float* __restrict__ rW2, const float* __restrict__ rb2,
                                float* __restrict__ out) {
    __shared__ float hs[256];
    const int b     = blockIdx.y;
    const int which = blockIdx.z;
    const float* W2 = which ? rW2 : vW2;
    const float* b2 = which ? rb2 : vb2;
    const float* h  = wsH + ((size_t)which * 64 + b) * 256;
    for (int k = threadIdx.x; k < 256; k += blockDim.x) hs[k] = h[k];
    __syncthreads();
    const int o = blockIdx.x * 256 + threadIdx.x;
    if (o < OUT_DIM) {
        float s = b2[o];
        for (int k = 0; k < 256; ++k)
            s = fmaf(hs[k], W2[(size_t)k * OUT_DIM + o], s);
        out[((size_t)which * 64 + b) * OUT_DIM + o] = s;
    }
}

__global__ void sat_kernel(const float* __restrict__ z,
                           const float* __restrict__ satW,
                           const float* __restrict__ satb,
                           float* __restrict__ policy) {
    const int b = threadIdx.x;                    // blockDim = 64
    float s = satb[0];
    for (int k = 0; k < LATENT; ++k)
        s = fmaf(z[(size_t)b * LATENT + k], satW[k], s);
    policy[(size_t)b * POLICY_DIM + NPAIR] = s;
}

// ---------------------------------------------------------------------------
// Launch
// ---------------------------------------------------------------------------
extern "C" void kernel_launch(void* const* d_in, const int* in_sizes, int n_in,
                              void* d_out, int out_size, void* d_ws, size_t ws_size,
                              hipStream_t stream) {
    (void)in_sizes; (void)n_in; (void)out_size; (void)ws_size;

    const float* z    = (const float*)d_in[0];
    const float* H    = (const float*)d_in[1];
    const float* vW1  = (const float*)d_in[2];
    const float* vb1  = (const float*)d_in[3];
    const float* vW2  = (const float*)d_in[4];
    const float* vb2  = (const float*)d_in[5];
    const float* rW1  = (const float*)d_in[6];
    const float* rb1  = (const float*)d_in[7];
    const float* rW2  = (const float*)d_in[8];
    const float* rb2  = (const float*)d_in[9];
    const float* sW1  = (const float*)d_in[10];
    const float* sb1  = (const float*)d_in[11];
    const float* sW2  = (const float*)d_in[12];
    const float* sb2  = (const float*)d_in[13];
    const float* satW = (const float*)d_in[14];
    const float* satb = (const float*)d_in[15];

    float* out    = (float*)d_out;
    float* v_out  = out;                                  // 64 x 601 (and r via 'which')
    float* policy = out + (size_t)2 * 64 * OUT_DIM;       // 64 x 8129

    char* ws = (char*)d_ws;
    unsigned* wsB = (unsigned*)(ws + 0);                  // 98304 B: packed bf16 sW1
    int*      iiT = (int*)(ws + 98304);                   // 32512 B
    int*      jjT = (int*)(ws + 130816);                  // 32512 B
    float*    wsH = (float*)(ws + 163328);                // 131072 B: head hiddens

    // one-time preps (cheap; re-run every call for determinism)
    pack_w1_kernel<<<dim3(96), dim3(256), 0, stream>>>(sW1, wsB);
    pair_table_kernel<<<dim3(32), dim3(256), 0, stream>>>(iiT, jjT);

    // hot kernel: 508 tiles x 64 batches, 4 waves/block
    edge_mlp_kernel<<<dim3(127, 64), dim3(128), 0, stream>>>(
        H, wsB, iiT, jjT, sb1, sW2, sb2, policy);

    // heads
    head_hidden_kernel<<<dim3(64, 2), dim3(256), 0, stream>>>(
        z, vW1, vb1, rW1, rb1, wsH);
    head_out_kernel<<<dim3(3, 64, 2), dim3(256), 0, stream>>>(
        wsH, vW2, vb2, rW2, rb2, v_out);
    sat_kernel<<<dim3(1), dim3(64), 0, stream>>>(z, satW, satb, policy);
}
